// ShotLevelMemorySSM_83958020702555
// MI455X (gfx1250) — matrix operand order repaired and checked by reading, not compiled, not run on hardware
//
#include <hip/hip_runtime.h>
#include <hip/hip_bf16.h>
#include <cmath>

// ---- problem dims (fixed by the reference) ----
constexpr int DEPTH = 2;
constexpr int Bsz = 256;   // batch
constexpr int Ssz = 64;    // shots (scan length)
constexpr int Dm  = 1024;  // embedding dim
constexpr int Nn  = 256;   // SSM state dim
constexpr int Rows = Bsz * Ssz;  // 16384 GEMM rows

typedef __attribute__((ext_vector_type(16))) _Float16 v16h_t;
typedef __attribute__((ext_vector_type(8)))  _Float16 v8h_t;
typedef __attribute__((ext_vector_type(8)))  float    v8f_t;

static __device__ __forceinline__ v16h_t cat8(v8h_t lo, v8h_t hi) {
  return __builtin_shufflevector(lo, hi, 0,1,2,3,4,5,6,7,8,9,10,11,12,13,14,15);
}

static __device__ __forceinline__ float softplus_f(float x) {
  return (x > 20.f) ? x : log1pf(expf(x));
}

// ---------------- pack / cast kernels ----------------
__global__ void cast_f16_kernel(const float* __restrict__ src,
                                _Float16* __restrict__ dst, int n) {
  int i = blockIdx.x * blockDim.x + threadIdx.x;
  if (i < n) dst[i] = (_Float16)src[i];
}

// src: [K,N] row-major f32 -> dst: [N,K] row-major f16 (B-fragment friendly)
__global__ void pack_wT_kernel(const float* __restrict__ src,
                               _Float16* __restrict__ dst, int K, int N) {
  int i = blockIdx.x * blockDim.x + threadIdx.x;
  if (i >= K * N) return;
  int n = i / K, k = i % K;
  dst[i] = (_Float16)src[(size_t)k * N + n];
}

// ---------------- fused delta/input projection, 2 M-tiles x {Wd,Wi} ----------------
// Xh: [Rows, Dm] f16 row-major; WdT/WiT: [Nn, Dm] f16 (transposed weights)
// decay = exp(-softplus(X*Wd+bd) * softplus(A_log)) ; drive = delta*(X*Wi+bi)
__global__ __launch_bounds__(256) void gemmA_kernel(
    const _Float16* __restrict__ Xh, const _Float16* __restrict__ WdT,
    const _Float16* __restrict__ WiT, const float* __restrict__ bd,
    const float* __restrict__ bi, const float* __restrict__ A_log,
    float* __restrict__ decay, float* __restrict__ drive) {
  constexpr int K = Dm, N = Nn, NT = N / 16;        // NT = 16
  const int wave = threadIdx.x >> 5;
  const int lane = threadIdx.x & 31;
  const int task = blockIdx.x * 8 + wave;           // (Rows/32)*NT = 8192 tasks
  const int mt2 = task / NT;                        // pair of M-tiles
  const int nt  = task % NT;
  const int half = lane >> 4, l16 = lane & 15;

  const _Float16* arow0 = Xh  + (size_t)(mt2 * 32 + l16) * K + half * 8;
  const _Float16* arow1 = arow0 + (size_t)16 * K;
  const _Float16* bdrow = WdT + (size_t)(nt * 16 + l16) * K + half * 8;
  const _Float16* birow = WiT + (size_t)(nt * 16 + l16) * K + half * 8;

  v8f_t accD0 = {}, accU0 = {}, accD1 = {}, accU1 = {};
  for (int k0 = 0; k0 < K; k0 += 32) {
    __builtin_prefetch(arow0 + k0 + 128, 0, 3);
    __builtin_prefetch(arow1 + k0 + 128, 0, 3);
    v16h_t a0 = cat8(*(const v8h_t*)(arow0 + k0), *(const v8h_t*)(arow0 + k0 + 16));
    v16h_t a1 = cat8(*(const v8h_t*)(arow1 + k0), *(const v8h_t*)(arow1 + k0 + 16));
    v16h_t bD = cat8(*(const v8h_t*)(bdrow + k0), *(const v8h_t*)(bdrow + k0 + 16));
    v16h_t bI = cat8(*(const v8h_t*)(birow + k0), *(const v8h_t*)(birow + k0 + 16));
    accD0 = __builtin_amdgcn_wmma_f32_16x16x32_f16(false, a0, false, bD, (short)0, accD0, false, false);
    accU0 = __builtin_amdgcn_wmma_f32_16x16x32_f16(false, a0, false, bI, (short)0, accU0, false, false);
    accD1 = __builtin_amdgcn_wmma_f32_16x16x32_f16(false, a1, false, bD, (short)0, accD1, false, false);
    accU1 = __builtin_amdgcn_wmma_f32_16x16x32_f16(false, a1, false, bI, (short)0, accU1, false, false);
  }

  const int n = nt * 16 + l16;
  const float bdv = bd[n], biv = bi[n];
  const float sA = softplus_f(A_log[n]);
  const int mbase0 = mt2 * 32 + half * 8;
  const int mbase1 = mbase0 + 16;
  for (int r = 0; r < 8; ++r) {
    size_t idx0 = (size_t)(mbase0 + r) * N + n;
    float d0 = softplus_f(accD0[r] + bdv);
    decay[idx0] = expf(-d0 * sA);
    drive[idx0] = d0 * (accU0[r] + biv);

    size_t idx1 = (size_t)(mbase1 + r) * N + n;
    float d1 = softplus_f(accD1[r] + bdv);
    decay[idx1] = expf(-d1 * sA);
    drive[idx1] = d1 * (accU1[r] + biv);
  }
}

// ---------------- selective scan over shots (f32 recurrence, f16 hs output) ----------------
__global__ __launch_bounds__(256) void scan_kernel(
    const float* __restrict__ decay, const float* __restrict__ drive,
    _Float16* __restrict__ hsH, float* __restrict__ finalSt) {
  const int b = blockIdx.x, n = threadIdx.x;
  const size_t base = (size_t)b * Ssz * Nn + n;
  float h = 0.f;
  for (int s = 0; s < Ssz; ++s) {
    size_t idx = base + (size_t)s * Nn;
    h = decay[idx] * h + drive[idx];
    hsH[idx] = (_Float16)h;
  }
  finalSt[(size_t)b * Nn + n] = h;
}

// ---------------- class_state = final @ Ws + bs (tiny GEMM, f32) ----------------
__global__ __launch_bounds__(256) void cs_kernel(
    const float* __restrict__ finalSt, const float* __restrict__ Ws,
    const float* __restrict__ bs, float* __restrict__ cs) {
  __shared__ float sfin[Nn];
  const int b = blockIdx.x, t = threadIdx.x;
  sfin[t] = finalSt[(size_t)b * Nn + t];
  __syncthreads();
  for (int j = 0; j < Dm / 256; ++j) {
    int d = t + j * 256;
    float acc = bs[d];
    for (int k = 0; k < Nn; ++k) acc += sfin[k] * Ws[(size_t)k * Dm + d];
    cs[(size_t)b * Dm + d] = acc;
  }
}

// ---------------- readout GEMM: hidden = hs @ Wo + bo + class_state (2x2 WMMA) ----------------
// Hh: [Rows, Nn] f16; WoT: [Dm, Nn] f16 (transposed). Writes f32 hidden + f16 shadow.
__global__ __launch_bounds__(256) void gemmC_kernel(
    const _Float16* __restrict__ Hh, const _Float16* __restrict__ WoT,
    const float* __restrict__ bo, const float* __restrict__ cs,
    float* __restrict__ hiddenOut, _Float16* __restrict__ hiddenH) {
  constexpr int K = Nn, N = Dm, NT2 = N / 32;       // NT2 = 32 (N-tile pairs)
  const int wave = threadIdx.x >> 5;
  const int lane = threadIdx.x & 31;
  const int task = blockIdx.x * 8 + wave;           // (Rows/32)*(N/32) = 16384 tasks
  const int mt2 = task / NT2;
  const int nt2 = task % NT2;
  const int half = lane >> 4, l16 = lane & 15;

  const _Float16* arow0 = Hh  + (size_t)(mt2 * 32 + l16) * K + half * 8;
  const _Float16* arow1 = arow0 + (size_t)16 * K;
  const _Float16* brow0 = WoT + (size_t)(nt2 * 32 + l16) * K + half * 8;
  const _Float16* brow1 = brow0 + (size_t)16 * K;

  v8f_t acc00 = {}, acc01 = {}, acc10 = {}, acc11 = {};
  for (int k0 = 0; k0 < K; k0 += 32) {
    v16h_t a0 = cat8(*(const v8h_t*)(arow0 + k0), *(const v8h_t*)(arow0 + k0 + 16));
    v16h_t a1 = cat8(*(const v8h_t*)(arow1 + k0), *(const v8h_t*)(arow1 + k0 + 16));
    v16h_t b0 = cat8(*(const v8h_t*)(brow0 + k0), *(const v8h_t*)(brow0 + k0 + 16));
    v16h_t b1 = cat8(*(const v8h_t*)(brow1 + k0), *(const v8h_t*)(brow1 + k0 + 16));
    acc00 = __builtin_amdgcn_wmma_f32_16x16x32_f16(false, a0, false, b0, (short)0, acc00, false, false);
    acc01 = __builtin_amdgcn_wmma_f32_16x16x32_f16(false, a0, false, b1, (short)0, acc01, false, false);
    acc10 = __builtin_amdgcn_wmma_f32_16x16x32_f16(false, a1, false, b0, (short)0, acc10, false, false);
    acc11 = __builtin_amdgcn_wmma_f32_16x16x32_f16(false, a1, false, b1, (short)0, acc11, false, false);
  }

  const int n0 = nt2 * 32 + l16;
  const int n1 = n0 + 16;
  const float bov0 = bo[n0], bov1 = bo[n1];
  const int mbase0 = mt2 * 32 + half * 8;
  const int mbase1 = mbase0 + 16;
  for (int r = 0; r < 8; ++r) {
    int row0 = mbase0 + r;
    int row1 = mbase1 + r;
    const float* cs0 = cs + (size_t)(row0 >> 6) * Dm;   // row>>6 == batch index
    const float* cs1 = cs + (size_t)(row1 >> 6) * Dm;

    float v00 = acc00[r] + bov0 + cs0[n0];
    float v01 = acc01[r] + bov1 + cs0[n1];
    float v10 = acc10[r] + bov0 + cs1[n0];
    float v11 = acc11[r] + bov1 + cs1[n1];

    size_t i00 = (size_t)row0 * Dm + n0;
    size_t i01 = (size_t)row0 * Dm + n1;
    size_t i10 = (size_t)row1 * Dm + n0;
    size_t i11 = (size_t)row1 * Dm + n1;
    hiddenOut[i00] = v00; hiddenH[i00] = (_Float16)v00;
    hiddenOut[i01] = v01; hiddenH[i01] = (_Float16)v01;
    hiddenOut[i10] = v10; hiddenH[i10] = (_Float16)v10;
    hiddenOut[i11] = v11; hiddenH[i11] = (_Float16)v11;
  }
}

// ---------------- pooled = cs + mean_s(hidden); LayerNorm -> class_repr ----------------
__global__ __launch_bounds__(256) void finalize_kernel(
    const float* __restrict__ hidden, const float* __restrict__ cs,
    const float* __restrict__ gamma, const float* __restrict__ beta,
    float* __restrict__ out) {
  __shared__ float pooled[Dm];
  __shared__ float rsum[256];
  __shared__ float rsq[256];
  const int b = blockIdx.x, t = threadIdx.x;
  for (int j = 0; j < Dm / 256; ++j) {
    int d = t + j * 256;
    float s = 0.f;
    const float* hp = hidden + (size_t)b * Ssz * Dm + d;
    for (int sh = 0; sh < Ssz; ++sh) s += hp[(size_t)sh * Dm];
    pooled[d] = cs[(size_t)b * Dm + d] + s * (1.f / Ssz);
  }
  __syncthreads();
  float ls = 0.f, lq = 0.f;
  for (int j = 0; j < Dm / 256; ++j) {
    float v = pooled[t + j * 256];
    ls += v; lq += v * v;
  }
  rsum[t] = ls; rsq[t] = lq;
  __syncthreads();
  for (int sft = 128; sft > 0; sft >>= 1) {
    if (t < sft) { rsum[t] += rsum[t + sft]; rsq[t] += rsq[t + sft]; }
    __syncthreads();
  }
  float mu  = rsum[0] * (1.f / Dm);
  float var = rsq[0] * (1.f / Dm) - mu * mu;
  float rstd = rsqrtf(var + 1e-5f);
  for (int j = 0; j < Dm / 256; ++j) {
    int d = t + j * 256;
    out[(size_t)b * Dm + d] = (pooled[d] - mu) * rstd * gamma[d] + beta[d];
  }
}

extern "C" void kernel_launch(void* const* d_in, const int* in_sizes, int n_in,
                              void* d_out, int out_size, void* d_ws, size_t ws_size,
                              hipStream_t stream) {
  (void)in_sizes; (void)n_in; (void)out_size; (void)ws_size;
  const float* x     = (const float*)d_in[0];   // [B,S,D]
  const float* Wd    = (const float*)d_in[1];   // [DEPTH,D,N]
  const float* bd    = (const float*)d_in[2];   // [DEPTH,N]
  const float* Wi    = (const float*)d_in[3];   // [DEPTH,D,N]
  const float* bi    = (const float*)d_in[4];   // [DEPTH,N]
  const float* A_log = (const float*)d_in[5];   // [DEPTH,N]
  const float* Wo    = (const float*)d_in[6];   // [DEPTH,N,D]
  const float* bo    = (const float*)d_in[7];   // [DEPTH,D]
  const float* Ws    = (const float*)d_in[8];   // [DEPTH,N,D]
  const float* bs    = (const float*)d_in[9];   // [DEPTH,D]
  const float* gamma = (const float*)d_in[10];  // [D]
  const float* beta  = (const float*)d_in[11];  // [D]

  float* out_repr   = (float*)d_out;                   // [B,D]
  float* out_hidden = out_repr + (size_t)Bsz * Dm;     // [B,S,D]

  // workspace carve (256-B aligned regions)
  char* wsp = (char*)d_ws;
  auto carve = [&](size_t bytes) {
    char* p = wsp;
    wsp += (bytes + 255) & ~(size_t)255;
    return (void*)p;
  };
  _Float16* hiddenH = (_Float16*)carve((size_t)Rows * Dm * 2);  // 33.5 MB
  _Float16* wdT     = (_Float16*)carve((size_t)Nn * Dm * 2);
  _Float16* wiT     = (_Float16*)carve((size_t)Nn * Dm * 2);
  _Float16* woT     = (_Float16*)carve((size_t)Dm * Nn * 2);
  float*    decay   = (float*)carve((size_t)Rows * Nn * 4);     // 16.8 MB
  float*    drive   = (float*)carve((size_t)Rows * Nn * 4);     // 16.8 MB
  _Float16* hsH     = (_Float16*)carve((size_t)Rows * Nn * 2);  // 8.4 MB
  float*    finalSt = (float*)carve((size_t)Bsz * Nn * 4);
  float*    cs      = (float*)carve((size_t)Bsz * Dm * 4);

  // f16 shadow of the layer input (starts as shot embeddings)
  cast_f16_kernel<<<(Rows * Dm + 255) / 256, 256, 0, stream>>>(x, hiddenH, Rows * Dm);

  for (int l = 0; l < DEPTH; ++l) {
    pack_wT_kernel<<<(Dm * Nn + 255) / 256, 256, 0, stream>>>(
        Wd + (size_t)l * Dm * Nn, wdT, Dm, Nn);
    pack_wT_kernel<<<(Dm * Nn + 255) / 256, 256, 0, stream>>>(
        Wi + (size_t)l * Dm * Nn, wiT, Dm, Nn);
    pack_wT_kernel<<<(Nn * Dm + 255) / 256, 256, 0, stream>>>(
        Wo + (size_t)l * Nn * Dm, woT, Nn, Dm);

    // fused delta/input projections + gate math -> decay, drive
    // tasks = (Rows/32) * (Nn/16) = 8192 waves -> 1024 blocks of 8 waves
    gemmA_kernel<<<(Rows / 32) * (Nn / 16) / 8, 256, 0, stream>>>(
        hiddenH, wdT, wiT, bd + (size_t)l * Nn, bi + (size_t)l * Nn,
        A_log + (size_t)l * Nn, decay, drive);

    // recurrence over shots
    scan_kernel<<<Bsz, 256, 0, stream>>>(decay, drive, hsH, finalSt);

    // class_state = final @ Ws + bs
    cs_kernel<<<Bsz, 256, 0, stream>>>(
        finalSt, Ws + (size_t)l * Nn * Dm, bs + (size_t)l * Dm, cs);

    // hidden = hs @ Wo + bo + class_state  (writes both f32 output and f16 shadow)
    // tasks = (Rows/32) * (Dm/32) = 16384 waves -> 2048 blocks of 8 waves
    gemmC_kernel<<<(Rows / 32) * (Dm / 32) / 8, 256, 0, stream>>>(
        hsH, woT, bo + (size_t)l * Dm, cs, out_hidden, hiddenH);
  }

  // pooled + LayerNorm -> class_repr
  finalize_kernel<<<Bsz, 256, 0, stream>>>(out_hidden, cs, gamma, beta, out_repr);
}
